// SpatialSelfCrossAttention_83691732730203
// MI455X (gfx1250) — compile-verified, matrix operand order checked by reading, not run on hardware
//
#include <hip/hip_runtime.h>

#define C_DIM 256
#define NPIX  2304
#define BATCH 8
#define USE_ASYNC 1

typedef __attribute__((ext_vector_type(16))) __bf16 v16bf;
typedef __attribute__((ext_vector_type(8)))  float  v8f;

union V16U {
  v16bf v;
  unsigned short u[16];
  unsigned int   w[8];
};

union Pack8 {
  unsigned short u[8];
  uint4 q;
};

__device__ __forceinline__ unsigned short f2bf(float f) {
  unsigned int u = __builtin_bit_cast(unsigned int, f);
  u += 0x7FFFu + ((u >> 16) & 1u);           // round-to-nearest-even
  return (unsigned short)(u >> 16);
}

// 16-bit A-matrix 16x32 K layout (CDNA5 ISA 7.12.2):
// VGPR v<4 -> K = 2v (+8 for lanes 16..31); v>=4 -> K = 16 + 2(v-4) (+8 hi)
// Both K runs {8hi..8hi+7} and {16+8hi..+7} are contiguous ushorts, so a
// row-major [M|N][K] tile loads as two ds_load_b128 per fragment.
__device__ __forceinline__ int kpair(int v, int hi) {
  return (v < 4 ? 2 * v : 8 + 2 * v) + hi * 8;
}

// Fragment loader for row-major [row][K] bf16 tiles. Used for A operands
// (row = M) and for B operands stored transposed (row = N).
__device__ __forceinline__ v16bf load_frag(const unsigned short* base, int rowStride,
                                           int row0, int lane) {
  const int m  = row0 + (lane & 15);
  const int hi = (lane >> 4) & 1;
  const unsigned short* r = base + (size_t)m * rowStride;
  V16U f;
#pragma unroll
  for (int v = 0; v < 8; ++v)
    f.w[v] = *reinterpret_cast<const unsigned int*>(r + kpair(v, hi));
  return f.v;
}

__device__ __forceinline__ v8f wmma_bf16(v16bf a, v16bf b, v8f c) {
  return __builtin_amdgcn_wmma_f32_16x16x32_bf16(false, a, false, b, (short)0, c,
                                                 false, false);
}

#if USE_ASYNC
// Async global -> LDS copy (16B per lane), tracked by ASYNCcnt.
__device__ __forceinline__ void async_ld16(const void* gptr, void* lptr) {
  const unsigned long long ga = reinterpret_cast<unsigned long long>(gptr);
  const unsigned lo = (unsigned)reinterpret_cast<unsigned long long>(lptr);
  asm volatile("global_load_async_to_lds_b128 %0, %1, off"
               :: "v"(lo), "v"(ga) : "memory");
}
__device__ __forceinline__ void async_wait() {
  asm volatile("s_wait_asynccnt 0x0" ::: "memory");
}
#endif

// ---------------------------------------------------------------------------
// GroupNorm (32 groups, 8 ch/group) -> bf16 normalized, TRANSPOSED [b][pix][C]
// ---------------------------------------------------------------------------
__global__ __launch_bounds__(256) void groupnorm_kernel(const float* __restrict__ x,
                                                        const float* __restrict__ gamma,
                                                        const float* __restrict__ beta,
                                                        unsigned short* __restrict__ outT) {
  const int b = blockIdx.x >> 5, g = blockIdx.x & 31;
  const int CPG = 8, GSZ = CPG * NPIX;                 // 18432 elements per group
  const float* xg = x + ((size_t)b * C_DIM + g * CPG) * NPIX;

  float s = 0.f, ss = 0.f;
  for (int idx = threadIdx.x; idx < GSZ; idx += 256) {
    float v = xg[idx]; s += v; ss += v * v;
  }
  __shared__ float red[256], red2[256];
  red[threadIdx.x] = s; red2[threadIdx.x] = ss;
  __syncthreads();
  for (int off = 128; off > 0; off >>= 1) {
    if (threadIdx.x < off) {
      red[threadIdx.x]  += red[threadIdx.x + off];
      red2[threadIdx.x] += red2[threadIdx.x + off];
    }
    __syncthreads();
  }
  __shared__ float sA[8], sB[8];
  if (threadIdx.x < 8) {
    const float mu  = red[0] * (1.0f / GSZ);
    const float var = red2[0] * (1.0f / GSZ) - mu * mu;
    const float rs  = rsqrtf(var + 1e-6f);
    const int   c   = g * CPG + threadIdx.x;
    const float ga  = gamma[c] * rs;
    sA[threadIdx.x] = ga;
    sB[threadIdx.x] = beta[c] - mu * ga;
  }
  __syncthreads();
  // transposed write: hT[b][pix][g*8 .. g*8+7], one 16B packed store per pixel
  unsigned short* og = outT + (size_t)b * NPIX * C_DIM + g * CPG;
  for (int p = threadIdx.x; p < NPIX; p += 256) {
    Pack8 pk;
#pragma unroll
    for (int ci = 0; ci < 8; ++ci)
      pk.u[ci] = f2bf(xg[(size_t)ci * NPIX + p] * sA[ci] + sB[ci]);
    *reinterpret_cast<uint4*>(og + (size_t)p * C_DIM) = pk.q;
  }
}

__global__ __launch_bounds__(256) void f32_to_bf16_kernel(const float* __restrict__ in,
                                                          unsigned short* __restrict__ out,
                                                          int n) {
  const int i = blockIdx.x * 256 + threadIdx.x;
  if (i < n) out[i] = f2bf(in[i]);
}

// ---------------------------------------------------------------------------
// QKV projections: out = W @ Hn + bias  (one wave per 32x32 output tile).
// Hn is transposed [b][n][C]; B tiles stage as [i][c] -> fast fragments.
// which: 0=q1T(scaled) 1=k1T 2=v(direct) 3=q2T(scaled) 4=k2T
// ---------------------------------------------------------------------------
__global__ __launch_bounds__(32) void qkv_proj_kernel(
    const unsigned short* __restrict__ h1T, const unsigned short* __restrict__ h2T,
    const unsigned short* __restrict__ wqb, const unsigned short* __restrict__ wkb,
    const unsigned short* __restrict__ wvb,
    const float* __restrict__ bq, const float* __restrict__ bk,
    const float* __restrict__ bv,
    unsigned short* __restrict__ q1T, unsigned short* __restrict__ k1T,
    unsigned short* __restrict__ vg,  unsigned short* __restrict__ q2T,
    unsigned short* __restrict__ k2T) {
  const int which = blockIdx.z % 5;
  const int b     = blockIdx.z / 5;
  const unsigned short* in = (which >= 3) ? h2T : h1T;
  const unsigned short* W  = (which == 0 || which == 3) ? wqb : (which == 2 ? wvb : wkb);
  const float* bias        = (which == 0 || which == 3) ? bq  : (which == 2 ? bv  : bk);
  unsigned short* out;
  bool  transposed;
  float scale = 1.0f;
  switch (which) {
    case 0:  out = q1T; transposed = true;  scale = 0.0625f; break;  // fold C^-0.5
    case 1:  out = k1T; transposed = true;  break;
    case 2:  out = vg;  transposed = false; break;
    case 3:  out = q2T; transposed = true;  scale = 0.0625f; break;
    default: out = k2T; transposed = true;  break;
  }

  __shared__ unsigned short sA[32 * 32];   // W  [o][c]
  __shared__ unsigned short sB[32 * 32];   // Hn [i][c]
  const int lane = threadIdx.x;
  const int oy = blockIdx.y * 32, ix = blockIdx.x * 32;
  v8f acc[2][2] = {};

  for (int c0 = 0; c0 < C_DIM; c0 += 32) {
    const uint4* sa = reinterpret_cast<const uint4*>(W + (size_t)(oy + lane) * C_DIM + c0);
    uint4* da = reinterpret_cast<uint4*>(sA + lane * 32);
    da[0] = sa[0]; da[1] = sa[1]; da[2] = sa[2]; da[3] = sa[3];
    const uint4* sb = reinterpret_cast<const uint4*>(
        in + ((size_t)b * NPIX + ix + lane) * C_DIM + c0);
    uint4* db = reinterpret_cast<uint4*>(sB + lane * 32);
    db[0] = sb[0]; db[1] = sb[1]; db[2] = sb[2]; db[3] = sb[3];
    __syncthreads();
    const v16bf a0 = load_frag(sA, 32, 0, lane);
    const v16bf a1 = load_frag(sA, 32, 16, lane);
#pragma unroll
    for (int ni = 0; ni < 2; ++ni) {
      const v16bf bf_ = load_frag(sB, 32, ni * 16, lane);
      acc[0][ni] = wmma_bf16(a0, bf_, acc[0][ni]);
      acc[1][ni] = wmma_bf16(a1, bf_, acc[1][ni]);
    }
    __syncthreads();
  }

  const int hi = lane >> 4, nl = lane & 15;
#pragma unroll
  for (int mi = 0; mi < 2; ++mi)
#pragma unroll
    for (int ni = 0; ni < 2; ++ni)
#pragma unroll
      for (int r = 0; r < 8; ++r) {
        const int oc = oy + mi * 16 + r + 8 * hi;   // D: M = r + 8*hi
        const int ip = ix + ni * 16 + nl;           // D: N = lane&15
        const float val = (acc[mi][ni][r] + bias[oc]) * scale;
        if (transposed)
          out[((size_t)b * NPIX + ip) * C_DIM + oc] = f2bf(val);
        else
          out[((size_t)b * C_DIM + oc) * NPIX + ip] = f2bf(val);
      }
}

// ---------------------------------------------------------------------------
// Fused dual-path flash attention (see round-0 notes). All LDS operand tiles
// are row-major-over-K so every fragment is 2x ds_load_b128.
// ---------------------------------------------------------------------------
__global__ __launch_bounds__(256, 1) void attn_kernel(
    const unsigned short* __restrict__ q1T, const unsigned short* __restrict__ q2T,
    const unsigned short* __restrict__ k1T, const unsigned short* __restrict__ k2T,
    const unsigned short* __restrict__ vg,
    unsigned short* __restrict__ hatT_self, unsigned short* __restrict__ hatT_cross) {
  const int b   = blockIdx.y;
  const int i0  = blockIdx.x * 32;
  const int tid = threadIdx.x;
  const int lane = tid & 31, wave = tid >> 5;
  const int aMat = wave & 1;            // phase A: 0 -> S1 (q1,k1), 1 -> S2
  const int aJh  = (wave >> 1) & 1;     // phase A: j half
  const int qh   = (wave >> 2) & 1;     // q half (phases A/B/C consistent)
  const int cq   = wave & 3;            // phase C: channel quarter (64 ch)
  const int sPath  = wave & 1;          // designated softmax path
  const bool isSoft = (wave == 0 || wave == 1 || wave == 4 || wave == 5);

  __shared__ unsigned short sKt1[32 * C_DIM];    // [j][c]  16 KB
  __shared__ unsigned short sKt2[32 * C_DIM];    // 16 KB
  __shared__ unsigned short sV[C_DIM * 32];      // [c][j]  16 KB
  __shared__ float          sS1[32 * 32];        // 4 KB
  __shared__ float          sS2[32 * 32];        // 4 KB
  __shared__ unsigned short sP[2][2][16 * 32];   // [path][qh][q][j] 4 KB
  __shared__ float          sScale[2][32];       // running rescale / inv-l

  // Preload this wave's Q fragments into registers (reused 72x)
  v16bf qfrag[8];
  {
    const unsigned short* qsrc =
        (aMat ? q2T : q1T) + ((size_t)b * NPIX + i0 + qh * 16) * C_DIM;
#pragma unroll
    for (int s = 0; s < 8; ++s) qfrag[s] = load_frag(qsrc + s * 32, C_DIM, 0, lane);
  }

  float m_row = -3.0e38f, l_row = 0.0f;   // meaningful only in designated waves
  v8f accO[2][4] = {};                    // [path][channel 16-tile] 16q x 64c

  const int krow = tid >> 3, kseg = tid & 7;     // K/Q-style tile staging split

  for (int j0 = 0; j0 < NPIX; j0 += 32) {
    __syncthreads();  // previous iteration's consumers done with LDS tiles

    // ---- stage K1^T, K2^T [j][c] and V [c][j] tiles -----------------------
    {
      const unsigned short* g1 = k1T + ((size_t)b * NPIX + j0 + krow) * C_DIM + kseg * 32;
      const unsigned short* g2 = k2T + ((size_t)b * NPIX + j0 + krow) * C_DIM + kseg * 32;
      const unsigned short* g3 = vg  + ((size_t)b * C_DIM + tid) * NPIX + j0;
      unsigned short* l1 = sKt1 + krow * C_DIM + kseg * 32;
      unsigned short* l2 = sKt2 + krow * C_DIM + kseg * 32;
      unsigned short* l3 = sV + tid * 32;
#if USE_ASYNC
#pragma unroll
      for (int s = 0; s < 4; ++s) {
        async_ld16(g1 + s * 8, l1 + s * 8);
        async_ld16(g2 + s * 8, l2 + s * 8);
        async_ld16(g3 + s * 8, l3 + s * 8);
      }
      async_wait();
#else
      const uint4* a1 = reinterpret_cast<const uint4*>(g1);
      const uint4* a2 = reinterpret_cast<const uint4*>(g2);
      const uint4* a3 = reinterpret_cast<const uint4*>(g3);
      uint4* d1 = reinterpret_cast<uint4*>(l1);
      uint4* d2 = reinterpret_cast<uint4*>(l2);
      uint4* d3 = reinterpret_cast<uint4*>(l3);
      d1[0] = a1[0]; d1[1] = a1[1]; d1[2] = a1[2]; d1[3] = a1[3];
      d2[0] = a2[0]; d2[1] = a2[1]; d2[2] = a2[2]; d2[3] = a2[3];
      d3[0] = a3[0]; d3[1] = a3[1]; d3[2] = a3[2]; d3[3] = a3[3];
#endif
      if (j0 + 32 < NPIX)  // warm caches for next j tile
        __builtin_prefetch(k1T + ((size_t)b * NPIX + j0 + 32 + krow) * C_DIM, 0, 0);
    }
    __syncthreads();

    // ---- Phase A: score tiles S = Q^T K (K-dim = C = 256) -----------------
    {
      const unsigned short* sKt = aMat ? sKt2 : sKt1;
      v8f sacc = {};
#pragma unroll
      for (int s = 0; s < 8; ++s) {
        const v16bf kf = load_frag(sKt + s * 32, C_DIM, aJh * 16, lane);
        sacc = wmma_bf16(qfrag[s], kf, sacc);
      }
      float* sSm = aMat ? sS2 : sS1;
      const int hi = lane >> 4, nl = lane & 15;
#pragma unroll
      for (int r = 0; r < 8; ++r)
        sSm[(qh * 16 + r + 8 * hi) * 32 + aJh * 16 + nl] = sacc[r];
    }
    __syncthreads();

    // ---- Phase B: online softmax (2 paths x 2 q-halves, lanes 0..15) ------
    if (isSoft && lane < 16) {
      const int q = qh * 16 + lane;
      const float* r1 = sS1 + q * 32;
      const float* r2 = sS2 + q * 32;
      float mt = -3.0e38f;
#pragma unroll
      for (int j = 0; j < 32; ++j) {
        const float sv = r1[j] + (sPath ? r2[j] : 0.0f);
        mt = fmaxf(mt, sv);
      }
      const float mnew = fmaxf(m_row, mt);
      const float scl  = __expf(m_row - mnew);
      float sum = 0.0f;
      unsigned short* pr = &sP[sPath][qh][lane * 32];
#pragma unroll
      for (int j = 0; j < 32; ++j) {
        const float sv = r1[j] + (sPath ? r2[j] : 0.0f);
        const float e  = __expf(sv - mnew);
        sum += e;
        pr[j] = f2bf(e);
      }
      l_row = l_row * scl + sum;
      m_row = mnew;
      sScale[sPath][q] = scl;
    }
    __syncthreads();

    // ---- Phase C: rescale accumulators, O += P @ V^T ----------------------
    {
      const int hi = lane >> 4;
#pragma unroll
      for (int p = 0; p < 2; ++p) {
        float sclr[8];
#pragma unroll
        for (int r = 0; r < 8; ++r) sclr[r] = sScale[p][qh * 16 + r + 8 * hi];
#pragma unroll
        for (int nt = 0; nt < 4; ++nt)
#pragma unroll
          for (int r = 0; r < 8; ++r) accO[p][nt][r] *= sclr[r];
        const v16bf pf = load_frag(&sP[p][qh][0], 32, 0, lane);
#pragma unroll
        for (int nt = 0; nt < 4; ++nt) {
          const v16bf vf = load_frag(sV, 32, cq * 64 + nt * 16, lane);
          accO[p][nt] = wmma_bf16(pf, vf, accO[p][nt]);
        }
      }
    }
  }

  // ---- finalize: divide by denominator, store transposed [b][n][C] bf16 ---
  __syncthreads();
  if (isSoft && lane < 16) sScale[sPath][qh * 16 + lane] = 1.0f / l_row;
  __syncthreads();
  {
    const int hi = lane >> 4, nl = lane & 15;
#pragma unroll
    for (int p = 0; p < 2; ++p) {
      unsigned short* out = p ? hatT_cross : hatT_self;
#pragma unroll
      for (int nt = 0; nt < 4; ++nt)
#pragma unroll
        for (int r = 0; r < 8; ++r) {
          const int qrow = qh * 16 + r + 8 * hi;
          const int c    = cq * 64 + nt * 16 + nl;
          const int ip   = i0 + qrow;
          out[((size_t)b * NPIX + ip) * C_DIM + c] =
              f2bf(accO[p][nt][r] * sScale[p][qrow]);
        }
    }
  }
}

// ---------------------------------------------------------------------------
// Output projection + bias + residual(x1): out = x1 + Wp @ h_att + bp (fp32)
// h_att is transposed [b][n][C]; B tiles stage as [i][c].
// ---------------------------------------------------------------------------
__global__ __launch_bounds__(32) void out_proj_kernel(
    const unsigned short* __restrict__ wpb,
    const unsigned short* __restrict__ hatT_self,
    const unsigned short* __restrict__ hatT_cross,
    const float* __restrict__ bp, const float* __restrict__ x1,
    float* __restrict__ dout) {
  const int b = blockIdx.z >> 1, path = blockIdx.z & 1;
  const unsigned short* in = path ? hatT_cross : hatT_self;
  float* out = dout + (size_t)path * ((size_t)BATCH * C_DIM * NPIX);

  __shared__ unsigned short sA[32 * 32];
  __shared__ unsigned short sB[32 * 32];
  const int lane = threadIdx.x;
  const int oy = blockIdx.y * 32, ix = blockIdx.x * 32;
  v8f acc[2][2] = {};

  for (int c0 = 0; c0 < C_DIM; c0 += 32) {
    const uint4* sa = reinterpret_cast<const uint4*>(wpb + (size_t)(oy + lane) * C_DIM + c0);
    uint4* da = reinterpret_cast<uint4*>(sA + lane * 32);
    da[0] = sa[0]; da[1] = sa[1]; da[2] = sa[2]; da[3] = sa[3];
    const uint4* sb = reinterpret_cast<const uint4*>(
        in + ((size_t)b * NPIX + ix + lane) * C_DIM + c0);
    uint4* db = reinterpret_cast<uint4*>(sB + lane * 32);
    db[0] = sb[0]; db[1] = sb[1]; db[2] = sb[2]; db[3] = sb[3];
    __syncthreads();
    const v16bf a0 = load_frag(sA, 32, 0, lane);
    const v16bf a1 = load_frag(sA, 32, 16, lane);
#pragma unroll
    for (int ni = 0; ni < 2; ++ni) {
      const v16bf bf_ = load_frag(sB, 32, ni * 16, lane);
      acc[0][ni] = wmma_bf16(a0, bf_, acc[0][ni]);
      acc[1][ni] = wmma_bf16(a1, bf_, acc[1][ni]);
    }
    __syncthreads();
  }

  const int hi = lane >> 4, nl = lane & 15;
#pragma unroll
  for (int mi = 0; mi < 2; ++mi)
#pragma unroll
    for (int ni = 0; ni < 2; ++ni)
#pragma unroll
      for (int r = 0; r < 8; ++r) {
        const int oc = oy + mi * 16 + r + 8 * hi;
        const int ip = ix + ni * 16 + nl;
        const size_t gidx = ((size_t)b * C_DIM + oc) * NPIX + ip;
        out[gidx] = acc[mi][ni][r] + bp[oc] + x1[gidx];
      }
}

// ---------------------------------------------------------------------------
extern "C" void kernel_launch(void* const* d_in, const int* in_sizes, int n_in,
                              void* d_out, int out_size, void* d_ws, size_t ws_size,
                              hipStream_t stream) {
  (void)in_sizes; (void)n_in; (void)out_size; (void)ws_size;
  const float* x1    = (const float*)d_in[0];
  const float* x2    = (const float*)d_in[1];
  const float* gamma = (const float*)d_in[2];
  const float* beta  = (const float*)d_in[3];
  const float* Wq    = (const float*)d_in[4];
  const float* bq    = (const float*)d_in[5];
  const float* Wk    = (const float*)d_in[6];
  const float* bk    = (const float*)d_in[7];
  const float* Wv    = (const float*)d_in[8];
  const float* bv    = (const float*)d_in[9];
  const float* Wp    = (const float*)d_in[10];
  const float* bp    = (const float*)d_in[11];

  char* ws = (char*)d_ws;
  size_t off = 0;
  auto take = [&](size_t bytes) -> char* {
    char* p = ws + off;
    off += (bytes + 255) & ~(size_t)255;
    return p;
  };
  const size_t TEN = (size_t)BATCH * C_DIM * NPIX * sizeof(unsigned short);
  unsigned short* h1T   = (unsigned short*)take(TEN);   // [b][n][C]
  unsigned short* h2T   = (unsigned short*)take(TEN);   // [b][n][C]
  unsigned short* q1T   = (unsigned short*)take(TEN);   // [b][n][C]
  unsigned short* q2T   = (unsigned short*)take(TEN);   // [b][n][C]
  unsigned short* k1T   = (unsigned short*)take(TEN);   // [b][n][C]
  unsigned short* k2T   = (unsigned short*)take(TEN);   // [b][n][C]
  unsigned short* vg    = (unsigned short*)take(TEN);   // [b][C][n]
  unsigned short* hat_s = (unsigned short*)take(TEN);   // [b][n][C]
  unsigned short* hat_c = (unsigned short*)take(TEN);   // [b][n][C]
  const size_t WSZ = (size_t)C_DIM * C_DIM * sizeof(unsigned short);
  unsigned short* wqb = (unsigned short*)take(WSZ);
  unsigned short* wkb = (unsigned short*)take(WSZ);
  unsigned short* wvb = (unsigned short*)take(WSZ);
  unsigned short* wpb = (unsigned short*)take(WSZ);

  groupnorm_kernel<<<dim3(BATCH * 32), 256, 0, stream>>>(x1, gamma, beta, h1T);
  groupnorm_kernel<<<dim3(BATCH * 32), 256, 0, stream>>>(x2, gamma, beta, h2T);

  f32_to_bf16_kernel<<<dim3(256), 256, 0, stream>>>(Wq, wqb, C_DIM * C_DIM);
  f32_to_bf16_kernel<<<dim3(256), 256, 0, stream>>>(Wk, wkb, C_DIM * C_DIM);
  f32_to_bf16_kernel<<<dim3(256), 256, 0, stream>>>(Wv, wvb, C_DIM * C_DIM);
  f32_to_bf16_kernel<<<dim3(256), 256, 0, stream>>>(Wp, wpb, C_DIM * C_DIM);

  qkv_proj_kernel<<<dim3(NPIX / 32, C_DIM / 32, BATCH * 5), 32, 0, stream>>>(
      h1T, h2T, wqb, wkb, wvb, bq, bk, bv, q1T, k1T, vg, q2T, k2T);

  attn_kernel<<<dim3(NPIX / 32, BATCH), 256, 0, stream>>>(
      q1T, q2T, k1T, k2T, vg, hat_s, hat_c);

  out_proj_kernel<<<dim3(NPIX / 32, C_DIM / 32, BATCH * 2), 32, 0, stream>>>(
      wpb, hat_s, hat_c, bp, x1, (float*)d_out);
}
